// CDE_41455024341485
// MI455X (gfx1250) — compile-verified, matrix operand order checked by reading, not run on hardware
//
#include <hip/hip_runtime.h>
#include <cstdint>
#include <cstddef>

// Problem constants (match reference)
#define BB 4096
#define TT 128
#define CC 7
#define EE 32
#define HH 128
#define CE 224          // C*E
#define NSTEP 16
#define NSTAGE 96       // 16 steps * 6 dopri5 stages

typedef __attribute__((ext_vector_type(16))) __bf16 v16bf;
typedef __attribute__((ext_vector_type(8)))  float  v8f;
typedef unsigned int u32x4 __attribute__((ext_vector_type(4)));
typedef int          i32x8 __attribute__((ext_vector_type(8)));
typedef int          i32x4 __attribute__((ext_vector_type(4)));

#if defined(__has_builtin)
#if __has_builtin(__builtin_amdgcn_tensor_load_to_lds) && \
    __has_builtin(__builtin_amdgcn_s_wait_tensorcnt)
#define HAVE_TDM 1
#endif
#endif

// ---------------------------------------------------------------------------
// Kernel 1: swizzle f32 weights [K][N] row-major -> bf16 WMMA B-fragment order.
// 16-bit B 32x16 fragment (mirror of A layout): lane n<16 holds col n,
// K elems {0..7,16..23}; lane n+16 holds col n, K elems {8..15,24..31};
// fragment element e <-> K = (e<8 ? e : 8+e) + (hi?8:0).
// Output: [n_tile][k_chunk][lane(32)][elem(16)] contiguous (32B per lane).
// ---------------------------------------------------------------------------
__global__ void wswz_kernel(const float* __restrict__ w, unsigned short* __restrict__ o,
                            int K, int N, int KC) {
  int i = blockIdx.x * blockDim.x + threadIdx.x;
  if (i >= K * N) return;
  int k = i / N, n = i % N;
  int kc = k >> 5, kl = k & 31;
  int hi = (kl >> 3) & 1;
  int rem = kl - hi * 8;                 // {0..7} U {16..23}
  int e = (rem < 8) ? rem : (rem - 8);
  int lane = (n & 15) + hi * 16;
  int nt = n >> 4;
  __bf16 bv = (__bf16)w[i];
  union { __bf16 h; unsigned short u; } cv; cv.h = bv;
  o[(((nt * KC + kc) * 32) + lane) * 16 + e] = cv.u;
}

// ---------------------------------------------------------------------------
// Kernel 2: natural cubic spline second derivatives M[b][t][c].
// t = arange(T) => h == 1, A = tridiag(1,4,1) of size T-2. Thomas per (b,c).
// ---------------------------------------------------------------------------
__global__ void spline_kernel(const float* __restrict__ x, float* __restrict__ Mbuf) {
  int tid = blockIdx.x * blockDim.x + threadIdx.x;
  if (tid >= BB * CC) return;
  int b = tid / CC, c = tid % CC;
  const float* xp = x + (size_t)b * (TT * CC) + c;

  float cp[126], dp[126];
  float xb = xp[CC];
  float sp = xb - xp[0];
  float cprev = 0.0f, dprev = 0.0f;
  for (int i = 0; i < 126; ++i) {
    float xc = xp[(i + 2) * CC];
    float sc = xc - xb;
    float rhs = 6.0f * (sc - sp);
    float inv = 1.0f / (4.0f - cprev);
    float cpi = inv;
    float dpi = (rhs - dprev) * inv;
    cp[i] = cpi; dp[i] = dpi;
    cprev = cpi; dprev = dpi;
    sp = sc; xb = xc;
  }
  float* Mp = Mbuf + (size_t)b * (TT * CC) + c;
  Mp[0] = 0.0f;
  Mp[(TT - 1) * CC] = 0.0f;
  float m = dp[125];
  Mp[126 * CC] = m;
  for (int i = 124; i >= 0; --i) {
    m = dp[i] - cp[i] * m;
    Mp[(i + 1) * CC] = m;
  }
}

// ---------------------------------------------------------------------------
// Kernel 3: evaluate dX/dt at the 96 dopri5 stage times -> dX[96][B][C].
// ---------------------------------------------------------------------------
__global__ void dxeval_kernel(const float* __restrict__ x, const float* __restrict__ Mbuf,
                              float* __restrict__ dX) {
  int i = blockIdx.x * blockDim.x + threadIdx.x;
  if (i >= NSTAGE * BB * CC) return;
  int j = i / (BB * CC);
  int r = i % (BB * CC);
  int b = r / CC, c = r % CC;
  int step = j / 6, sI = j % 6;
  const float off[6] = {0.0f, 0.2f, 0.3f, 0.8f, 8.0f / 9.0f, 1.0f};
  const float dt = 127.0f / 16.0f;
  float ti = ((float)step + off[sI]) * dt;
  int idx = (int)floorf(ti);
  idx = idx < 0 ? 0 : (idx > 126 ? 126 : idx);
  float s = ti - (float)idx;
  size_t base = ((size_t)b * TT + idx) * CC + c;
  float x0 = x[base], x1 = x[base + CC];
  float M0 = Mbuf[base], M1 = Mbuf[base + CC];
  float slope = x1 - x0;
  float bco = slope - (2.0f * M0 + M1) * (1.0f / 6.0f);
  float dco = (M1 - M0) * (1.0f / 6.0f);
  dX[i] = bco + M0 * s + 3.0f * dco * s * s;
}

// ---------------------------------------------------------------------------
// Fused ODE kernel helpers
// ---------------------------------------------------------------------------
__device__ __forceinline__ v16bf ld_bfrag(const __bf16* p) {  // contiguous 16 halves
  union { v16bf v; uint4 u[2]; } U;
  U.u[0] = *(const uint4*)(p);
  U.u[1] = *(const uint4*)(p + 8);
  return U.v;
}
__device__ __forceinline__ v16bf ld_afrag(const __bf16* p) {  // 8 halves, gap 16, 8 halves
  union { v16bf v; uint4 u[2]; } U;
  U.u[0] = *(const uint4*)(p);
  U.u[1] = *(const uint4*)(p + 16);
  return U.v;
}

__device__ __forceinline__ float act_fn(float v, bool do_tanh) {
  return do_tanh ? tanhf(v) : ((v > 0.0f) ? v : 0.0f);
}

// nt loop unrolled x2: two independent accumulator chains so one tile's
// D->VALU consumption hides behind the other tile's WMMA chain.
template<int KC, int NT, bool DO_TANH>
__device__ __forceinline__ void mlp_layer(const v16bf (&af)[KC],
                                          const __bf16* wl, const float* bl,
                                          __bf16* actw, int lane, int ln16, int hi) {
  static_assert(NT % 2 == 0, "NT must be even");
  #pragma unroll 1
  for (int nt = 0; nt < NT; nt += 2) {
    float bva = bl[nt * 16 + ln16];
    float bvb = bl[nt * 16 + 16 + ln16];
    v8f acca = {bva, bva, bva, bva, bva, bva, bva, bva};
    v8f accb = {bvb, bvb, bvb, bvb, bvb, bvb, bvb, bvb};
    #pragma unroll
    for (int kc = 0; kc < KC; ++kc) {
      v16bf bfa = ld_bfrag(wl + ((nt * KC + kc) * 32 + lane) * 16);
      v16bf bfb = ld_bfrag(wl + (((nt + 1) * KC + kc) * 32 + lane) * 16);
      acca = __builtin_amdgcn_wmma_f32_16x16x32_bf16(false, af[kc], false, bfa,
                                                     (short)0, acca, false, false);
      accb = __builtin_amdgcn_wmma_f32_16x16x32_bf16(false, af[kc], false, bfb,
                                                     (short)0, accb, false, false);
    }
    // C/D layout: VGPR r -> row r + 8*hi, col = nt*16 + ln16
    #pragma unroll
    for (int r = 0; r < 8; ++r) {
      float va = act_fn(acca[r], DO_TANH);
      float vb = act_fn(accb[r], DO_TANH);
      actw[(r + hi * 8) * CE + nt * 16 + ln16] = (__bf16)va;
      actw[(r + hi * 8) * CE + nt * 16 + 16 + ln16] = (__bf16)vb;
    }
  }
}

// ---------------------------------------------------------------------------
// Kernel 4: fully fused 16-step dopri5 CDE integrator.
// 64 rows per block, 4 waves, wave-private 16-row tile. Weights in LDS (bf16,
// DMA'd in by the Tensor Data Mover), z/k ODE state register-resident in WMMA
// A-fragment lane layout.
// ---------------------------------------------------------------------------
__global__ void __launch_bounds__(128, 1)
cde_main_kernel(const float* __restrict__ x,
                const float* __restrict__ embw, const float* __restrict__ embb,
                const float* __restrict__ b0g, const float* __restrict__ b1g,
                const float* __restrict__ b2g, const float* __restrict__ b3g,
                const unsigned short* __restrict__ wsw,   // [w0|w1|w2|w3] swizzled bf16
                const float* __restrict__ dXg,
                float* __restrict__ outg) {
  extern __shared__ uint4 smem_u4[];
  __bf16* smh = (__bf16*)smem_u4;
  __bf16* w0l = smh;              // 32x128  -> 4096 halves
  __bf16* w1l = smh + 4096;       // 128x128 -> 16384
  __bf16* w2l = smh + 20480;      // 128x128 -> 16384
  __bf16* w3l = smh + 36864;      // 128x224 -> 28672
  float*  bl  = (float*)(smh + 65536);     // 608 floats (b0|b1|b2|b3)
  __bf16* act = (__bf16*)(bl + 608);       // 4 waves * 16 * 224 bf16

  const int tid = threadIdx.x;
  const int wave = tid >> 5, lane = tid & 31;

#ifdef HAVE_TDM
  // TDM: one DMA moves the 131072-byte swizzled weight block into LDS offset 0.
  // D# 2D descriptor: data_size=8B, tile_dim0=16384, tile_dim1=1 row.
  if (wave == 0) {
    unsigned long long ga = (unsigned long long)(const void*)wsw;
    u32x4 g0;
    g0.x = 1u;                                   // count=1, user mode, no gather
    g0.y = 0u;                                   // lds_addr = 0 (weight region)
    g0.z = (unsigned)(ga & 0xffffffffu);         // global_addr[31:0]
    g0.w = (unsigned)((ga >> 32) & 0x1ffffffu)   // global_addr[56:32]
           | (2u << 30);                         // type = 2 ("image")
    i32x8 g1;
    g1[0] = (3 << 16);                           // wg_mask=0, data_size=8B
    g1[1] = (int)(16384u << 16);                 // tensor_dim0[15:0] @ bits 63:48
    g1[2] = (16384 >> 16) | (1 << 16);           // tensor_dim0[31:16] | tensor_dim1 lo
    g1[3] = (int)(16384u << 16);                 // tensor_dim1 hi=0 | tile_dim0 @127:112
    g1[4] = 1;                                   // tile_dim1=1, tile_dim2=0
    g1[5] = 16384;                               // tensor_dim0_stride[31:0]
    g1[6] = (int)(16384u << 16);                 // stride0 hi=0 | tensor_dim1_stride lo
    g1[7] = 16384 >> 16;                         // tensor_dim1_stride hi
    i32x4 gz = {0, 0, 0, 0};
#if defined(__clang_major__) && (__clang_major__ >= 23)
    i32x8 gz8 = {0, 0, 0, 0, 0, 0, 0, 0};
    __builtin_amdgcn_tensor_load_to_lds(g0, g1, gz, gz, gz8, 0);
#else
    __builtin_amdgcn_tensor_load_to_lds(g0, g1, gz, gz, 0);
#endif
    __builtin_amdgcn_s_wait_tensorcnt(0);
  }
#else
  const uint4* ws4 = (const uint4*)wsw;
  for (int i = tid; i < 8192; i += 128) smem_u4[i] = ws4[i];
#endif
  for (int i = tid; i < 608; i += 128) {
    float v;
    if      (i < 128) v = b0g[i];
    else if (i < 256) v = b1g[i - 128];
    else if (i < 384) v = b2g[i - 256];
    else              v = b3g[i - 384];
    bl[i] = v;
  }
  __syncthreads();

  const int ln16 = lane & 15, hi = lane >> 4;
  const int myrow = blockIdx.x * 64 + wave * 16 + ln16;
  __bf16* actw = act + wave * (16 * CE);

  // A-fragment column map: col(j) = (j<8 ? j : j+8) + 8*hi
  float z[16];
  {
    float xr[CC];
    #pragma unroll
    for (int c = 0; c < CC; ++c) xr[c] = x[(size_t)myrow * (TT * CC) + c];  // x[:,0,:]
    #pragma unroll
    for (int j = 0; j < 16; ++j) {
      int e = ((j < 8) ? j : j + 8) + hi * 8;
      float a = embb[e];
      #pragma unroll
      for (int c = 0; c < CC; ++c) a += xr[c] * embw[c * EE + e];
      z[j] = a;
    }
  }

  float kk[6][16];
  const float dt = 127.0f / 16.0f;
  constexpr float STC[6][5] = {
    {0, 0, 0, 0, 0},
    {0.2f, 0, 0, 0, 0},
    {3.f/40.f, 9.f/40.f, 0, 0, 0},
    {44.f/45.f, -56.f/15.f, 32.f/9.f, 0, 0},
    {19372.f/6561.f, -25360.f/2187.f, 64448.f/6561.f, -212.f/729.f, 0},
    {9017.f/3168.f, -355.f/33.f, 46732.f/5247.f, 49.f/176.f, -5103.f/18656.f}
  };
  constexpr float FIN[6] = {35.f/384.f, 0.f, 500.f/1113.f, 125.f/192.f,
                            -2187.f/6784.f, 11.f/84.f};

  for (int step = 0; step < NSTEP; ++step) {
    #pragma unroll
    for (int s = 0; s < 6; ++s) {
      // Stage input zst = z + dt * sum_q STC[s][q] * k_q, built directly as A-frag
      v16bf a0;
      #pragma unroll
      for (int j = 0; j < 16; ++j) {
        float v = z[j];
        #pragma unroll
        for (int q = 0; q < 5; ++q)
          if (q < s) v += (dt * STC[s][q]) * kk[q][j];
        a0[j] = (__bf16)v;
      }
      v16bf a1arr[1] = {a0};
      mlp_layer<1, 8, false>(a1arr, w0l, bl, actw, lane, ln16, hi);        // 32 -> 128

      v16bf af[4];
      #pragma unroll
      for (int kc = 0; kc < 4; ++kc)
        af[kc] = ld_afrag(actw + ln16 * CE + kc * 32 + hi * 8);
      mlp_layer<4, 8, false>(af, w1l, bl + 128, actw, lane, ln16, hi);     // 128 -> 128
      #pragma unroll
      for (int kc = 0; kc < 4; ++kc)
        af[kc] = ld_afrag(actw + ln16 * CE + kc * 32 + hi * 8);
      mlp_layer<4, 8, false>(af, w2l, bl + 256, actw, lane, ln16, hi);     // 128 -> 128
      #pragma unroll
      for (int kc = 0; kc < 4; ++kc)
        af[kc] = ld_afrag(actw + ln16 * CE + kc * 32 + hi * 8);
      mlp_layer<4, 14, true>(af, w3l, bl + 384, actw, lane, ln16, hi);     // 128 -> 224

      // k_s[row][e] = sum_c F[row][e*7+c] * dXdt[row][c]
      const int stg = step * 6 + s;
      float dxr[CC];
      #pragma unroll
      for (int c = 0; c < CC; ++c)
        dxr[c] = dXg[((size_t)stg * BB + myrow) * CC + c];
      if (stg + 1 < NSTAGE)
        __builtin_prefetch(&dXg[((size_t)(stg + 1) * BB + myrow) * CC], 0, 3);
      #pragma unroll
      for (int j = 0; j < 16; ++j) {
        int e = ((j < 8) ? j : j + 8) + hi * 8;
        float a = 0.0f;
        #pragma unroll
        for (int c = 0; c < CC; ++c)
          a += (float)actw[ln16 * CE + e * CC + c] * dxr[c];
        kk[s][j] = a;
      }
    }
    // dopri5 5th-order update
    #pragma unroll
    for (int j = 0; j < 16; ++j)
      z[j] += dt * (FIN[0] * kk[0][j] + FIN[2] * kk[2][j] + FIN[3] * kk[3][j] +
                    FIN[4] * kk[4][j] + FIN[5] * kk[5][j]);
  }

  #pragma unroll
  for (int j = 0; j < 16; ++j) {
    int e = ((j < 8) ? j : j + 8) + hi * 8;
    outg[(size_t)myrow * EE + e] = z[j];
  }
}

// ---------------------------------------------------------------------------
extern "C" void kernel_launch(void* const* d_in, const int* in_sizes, int n_in,
                              void* d_out, int out_size, void* d_ws, size_t ws_size,
                              hipStream_t stream) {
  (void)in_sizes; (void)n_in; (void)out_size; (void)ws_size;
  const float* x    = (const float*)d_in[1];
  const float* embw = (const float*)d_in[2];
  const float* embb = (const float*)d_in[3];
  const float* w0   = (const float*)d_in[4];
  const float* b0   = (const float*)d_in[5];
  const float* w1   = (const float*)d_in[6];
  const float* b1   = (const float*)d_in[7];
  const float* w2   = (const float*)d_in[8];
  const float* b2   = (const float*)d_in[9];
  const float* w3   = (const float*)d_in[10];
  const float* b3   = (const float*)d_in[11];
  float* out = (float*)d_out;

  char* ws = (char*)d_ws;
  float* Mbuf = (float*)ws;                                     // B*T*C f32
  float* dX   = (float*)(ws + (size_t)BB * TT * CC * 4);        // 96*B*C f32
  unsigned short* wsw =
      (unsigned short*)(ws + (size_t)BB * TT * CC * 4 + (size_t)NSTAGE * BB * CC * 4);

  // Weight swizzle (f32 -> bf16 B-fragment layout), halves offsets: 0,4096,20480,36864
  wswz_kernel<<<(32 * 128 + 255) / 256, 256, 0, stream>>>(w0, wsw + 0,     32, 128, 1);
  wswz_kernel<<<(128 * 128 + 255) / 256, 256, 0, stream>>>(w1, wsw + 4096, 128, 128, 4);
  wswz_kernel<<<(128 * 128 + 255) / 256, 256, 0, stream>>>(w2, wsw + 20480, 128, 128, 4);
  wswz_kernel<<<(128 * 224 + 255) / 256, 256, 0, stream>>>(w3, wsw + 36864, 128, 224, 4);

  spline_kernel<<<(BB * CC + 127) / 128, 128, 0, stream>>>(x, Mbuf);
  dxeval_kernel<<<(NSTAGE * BB * CC + 255) / 256, 256, 0, stream>>>(x, Mbuf, dX);

  const int SMEM = 131072 + 2432 + 4 * 16 * CE * 2;  // weights + biases + act = 162176 B
  (void)hipFuncSetAttribute((const void*)cde_main_kernel,
                            hipFuncAttributeMaxDynamicSharedMemorySize, SMEM);
  cde_main_kernel<<<BB / 64, 128, SMEM, stream>>>(x, embw, embb, b0, b1, b2, b3,
                                                  wsw, dX, out);
}
